// HSIKeyBankAttention_18193481465904
// MI455X (gfx1250) — matrix operand order repaired and checked by reading, not compiled, not run on hardware
//
#include <hip/hip_runtime.h>
#include <hip/hip_bf16.h>
#include <math.h>

// ---------------- problem constants ----------------
#define B_   8
#define N_   4096
#define C_   512
#define H_   8
#define D_   64
#define M_   16
#define BN_  (B_ * N_)          // 32768 token rows
#define TAU_INV 10.0f

// ---------------- CDNA5 WMMA types ----------------
typedef _Float16 v16h __attribute__((ext_vector_type(16)));
typedef _Float16 v8h  __attribute__((ext_vector_type(8)));
typedef float    v8f  __attribute__((ext_vector_type(8)));
typedef float    v4f  __attribute__((ext_vector_type(4)));

#define WMMA_F16(a, b, c) \
  __builtin_amdgcn_wmma_f32_16x16x32_f16(false, (a), false, (b), (short)0, (c), false, false)

// Fragment loaders from LDS tiles staged as [row][32] halves (64B row stride).
// A 16x32 f16 (ISA 7.12.2): lane<16 -> M=lane, halves = K0..7 | K16..23 ;
// lane>=16 -> M=lane-16, halves = K8..15 | K24..31.
__device__ inline v16h frag_a(const _Float16 S[][32], int row, int lane) {
  const _Float16* p = &S[row][0];
  int base = (lane & 16) ? 8 : 0;
  v8h lo = *(const v8h*)(p + base);
  v8h hi = *(const v8h*)(p + base + 16);
  v16h r;
#pragma unroll
  for (int i = 0; i < 8; ++i) { r[i] = lo[i]; r[i + 8] = hi[i]; }
  return r;
}
// B 32x16 f16: lane<16 -> N=lane, K0..7 | K16..23 ; lane>=16 -> N=lane-16, K8..15 | K24..31.
// Tile staged column-major: S[col][k] so each lane reads two runs of 8 contiguous halves.
__device__ inline v16h frag_b(const _Float16 S[][32], int col, int lane) {
  const _Float16* p = &S[col][0];
  int base = (lane & 16) ? 8 : 0;
  v8h lo = *(const v8h*)(p + base);
  v8h hi = *(const v8h*)(p + base + 16);
  v16h r;
#pragma unroll
  for (int i = 0; i < 8; ++i) { r[i] = lo[i]; r[i + 8] = hi[i]; }
  return r;
}

__device__ inline v8h cvt8(v4f a, v4f b) {
  v8h h;
#pragma unroll
  for (int j = 0; j < 4; ++j) { h[j] = (_Float16)a[j]; h[4 + j] = (_Float16)b[j]; }
  return h;
}

// ---------------- generic tiled WMMA GEMM ----------------
// C[z] = A[z] * B[z] (+bias). f32 in/out, f16 WMMA math, f32 accumulate.
// Preconditions used by all call sites: M % 64 == 0, K % 32 == 0, all row
// strides multiples of 4 floats (16B vector loads legal). Only N may be
// ragged (sims GEMM) -> compile-time NGUARD.
template <bool TRANSB, bool NGUARD>
__global__ __launch_bounds__(256) void gemm_wmma(
    const float* __restrict__ A, const float* __restrict__ Bm,
    float* __restrict__ Cm, const float* __restrict__ bias,
    int M, int N, int K, int lda, int ldb, int ldc,
    long a1, long a2, long b1, long b2, long c1, long c2, int zdiv)
{
  __shared__ alignas(16) _Float16 sA[64][32];
  __shared__ alignas(16) _Float16 sB[64][32];
  int z = blockIdx.z, z1 = z / zdiv, z2 = z % zdiv;
  A  += z1 * a1 + z2 * a2;
  Bm += z1 * b1 + z2 * b2;
  Cm += z1 * c1 + z2 * c2;
  int m0 = blockIdx.y * 64, n0 = blockIdx.x * 64;
  int t = threadIdx.x, lane = t & 31, wave = t >> 5;
  int rw = wave & 3, cw = wave >> 2;            // 4 row-groups x 2 col-groups

  // A staging map: 8 contiguous K per thread -> one 16B LDS store
  const int am = t >> 2, ak = (t & 3) * 8;
  const float* aptr = A + (long)(m0 + am) * lda + ak;
  // B staging maps
  const int bn_t = t >> 2, bk_t = (t & 3) * 8;  // TRANSB: row n, 8 contiguous k
  const int bk_n = t >> 3, bn_n = (t & 7) * 8;  // normal: row k, 8 contiguous n

  v8f acc0 = {}; v8f acc1 = {};
  for (int k0 = 0; k0 < K; k0 += 32) {
    { // ---- stage A 64x32 (vectorized, no guards needed) ----
      v4f a0 = *(const v4f*)(aptr + k0);
      v4f a1v = *(const v4f*)(aptr + k0 + 4);
      *(v8h*)&sA[am][ak] = cvt8(a0, a1v);
      if (k0 + 32 < K) __builtin_prefetch(aptr + k0 + 32, 0, 1);
    }
    if (TRANSB) { // ---- B given (N,K): contiguous k, optional column guard ----
      v8h hv = {};
      int gn = n0 + bn_t;
      if (!NGUARD || gn < N) {
        const float* bp = Bm + (long)gn * ldb + k0 + bk_t;
        hv = cvt8(*(const v4f*)bp, *(const v4f*)(bp + 4));
      }
      *(v8h*)&sB[bn_t][bk_t] = hv;
    } else {      // ---- B given (K,N): vector load along n, transposed stores ----
      const float* bp = Bm + (long)(k0 + bk_n) * ldb + n0 + bn_n;
      v4f b0 = *(const v4f*)bp;
      v4f b1v = *(const v4f*)(bp + 4);
#pragma unroll
      for (int j = 0; j < 4; ++j) {
        sB[bn_n + j][bk_n]     = (_Float16)b0[j];
        sB[bn_n + 4 + j][bk_n] = (_Float16)b1v[j];
      }
    }
    __syncthreads();
    v16h a  = frag_a(sA, rw * 16 + (lane & 15), lane);
    v16h f0 = frag_b(sB, cw * 32 + (lane & 15), lane);
    v16h f1 = frag_b(sB, cw * 32 + 16 + (lane & 15), lane);
    acc0 = WMMA_F16(a, f0, acc0);
    acc1 = WMMA_F16(a, f1, acc1);
    __syncthreads();
  }
  // D layout: lane<16 -> N=lane, M=r ; lane>=16 -> N=lane-16, M=8+r
  int nloc = lane & 15, mb = (lane < 16) ? 0 : 8;
#pragma unroll
  for (int r = 0; r < 8; ++r) {
    int gm = m0 + rw * 16 + mb + r;          // M is always a multiple of 64
    int gn = n0 + cw * 32 + nloc;
    if (!NGUARD || gn < N)      Cm[(long)gm * ldc + gn]      = acc0[r] + (bias ? bias[gn] : 0.f);
    if (!NGUARD || gn + 16 < N) Cm[(long)gm * ldc + gn + 16] = acc1[r] + (bias ? bias[gn + 16] : 0.f);
  }
}

// ---------------- top-3 routing (spec+spat fused, writes concatenated r) ----
__global__ __launch_bounds__(256) void routing_kernel(
    const float* __restrict__ lgspec, const float* __restrict__ lgspat,
    float* __restrict__ rcat)
{
  long idx = (long)blockIdx.x * 256 + threadIdx.x;   // over H_*BN_
  if (idx >= (long)H_ * BN_) return;
  int  h  = (int)(idx >> 15);                        // BN_ == 1<<15
  long bn = idx & (BN_ - 1);
  float* dstbase = rcat + ((long)h * BN_ + bn) * 32;
#pragma unroll
  for (int which = 0; which < 2; ++which) {
    const float* lg = (which ? lgspat : lgspec) + bn * (H_ * M_) + h * M_;
    float l[M_];
#pragma unroll
    for (int q = 0; q < 4; ++q) {
      v4f v = *(const v4f*)(lg + 4 * q);
#pragma unroll
      for (int j = 0; j < 4; ++j) l[4 * q + j] = v[j];
    }
    float m1 = -1e30f, m2 = -1e30f, m3 = -1e30f;
#pragma unroll
    for (int m = 0; m < M_; ++m) {
      float v = l[m];
      if (v > m1)      { m3 = m2; m2 = m1; m1 = v; }
      else if (v > m2) { m3 = m2; m2 = v; }
      else if (v > m3) { m3 = v; }
    }
    float se = 0.f;
#pragma unroll
    for (int m = 0; m < M_; ++m) se += expf(l[m] - m1);
    float inv = 1.f / se;
    float ps[M_]; float dm = 0.f;
#pragma unroll
    for (int m = 0; m < M_; ++m) {
      float p = (l[m] >= m3) ? expf(l[m] - m1) * inv : 0.f;
      ps[m] = p; dm += p;
    }
    float rden = 1.f / fmaxf(dm, 1e-6f);
    float* dst = dstbase + which * 16;
#pragma unroll
    for (int q = 0; q < 4; ++q) {
      v4f v;
#pragma unroll
      for (int j = 0; j < 4; ++j) v[j] = ps[4 * q + j] * rden;
      *(v4f*)(dst + 4 * q) = v;
    }
  }
}

// ---------------- concat two (H,16,64) banks -> (H,32,64) ----------------
__global__ __launch_bounds__(256) void concat_kernel(
    const float* __restrict__ P0, const float* __restrict__ P1,
    float* __restrict__ dst)
{
  int idx = blockIdx.x * 256 + threadIdx.x;          // H_*32*64 = 16384
  if (idx >= H_ * 32 * 64) return;
  int h = idx >> 11, r = (idx >> 6) & 31, d = idx & 63;
  dst[idx] = (r < 16) ? P0[((long)h * 16 + r) * 64 + d]
                      : P1[((long)h * 16 + (r - 16)) * 64 + d];
}

// ---------------- column softmax stats over N (online max/sum) -------------
__global__ __launch_bounds__(256) void softmax_stats(
    const float* __restrict__ keff, float* __restrict__ mx, float* __restrict__ sm)
{
  int hb = blockIdx.x;                               // h*B_ + b
  int h = hb >> 3, b = hb & 7;
  long base = ((long)h * BN_ + (long)b * N_) * 64;
  int t = threadIdx.x, d = t & 63, seg = t >> 6;     // 4 segments of 1024 rows
  float m = -1e30f, s = 0.f;
  for (int n = seg * 1024; n < (seg + 1) * 1024; ++n) {
    float v = keff[base + (long)n * 64 + d];
    if (v > m) { s = s * expf(m - v) + 1.f; m = v; }
    else         s += expf(v - m);
  }
  __shared__ float sm_m[256], sm_s[256];
  sm_m[t] = m; sm_s[t] = s;
  __syncthreads();
  if (t < 64) {
    float M0 = sm_m[t], S0 = sm_s[t];
#pragma unroll
    for (int g = 1; g < 4; ++g) {
      float M1 = sm_m[g * 64 + t], S1 = sm_s[g * 64 + t];
      if (M1 > M0) { S0 = S0 * expf(M0 - M1) + S1; M0 = M1; }
      else           S0 += S1 * expf(M1 - M0);
    }
    mx[hb * 64 + t] = M0; sm[hb * 64 + t] = S0;
  }
}

// ---------------- context = K_norm^T . v  (64x64 per (h,b), WMMA) ----------
__global__ __launch_bounds__(256) void context_wmma(
    const float* __restrict__ keff, const float* __restrict__ vbuf,
    const float* __restrict__ mx, const float* __restrict__ sm,
    float* __restrict__ ctx)
{
  int hb = blockIdx.x; int h = hb >> 3, b = hb & 7;
  long kbase = ((long)h * BN_ + (long)b * N_) * 64;
  long vbase = ((long)b * N_) * C_ + (long)h * 64;
  __shared__ alignas(16) _Float16 sK[64][32];        // [d][n] = K_norm^T tile
  __shared__ alignas(16) _Float16 sV[64][32];        // [e][n] = v^T tile
  __shared__ float smax[64], sinv[64];
  int t = threadIdx.x, lane = t & 31, wave = t >> 5;
  if (t < 64) { smax[t] = mx[hb * 64 + t]; sinv[t] = 1.f / sm[hb * 64 + t]; }
  __syncthreads();
  int rw = wave & 3, cw = wave >> 2;
  // staging map: row n = t>>3 (0..31), 8 contiguous d per thread (coalesced)
  const int sn = t >> 3, sd = (t & 7) * 8;
  v8f acc0 = {}; v8f acc1 = {};
  for (int n0 = 0; n0 < N_; n0 += 32) {
    const float* kp = keff + kbase + (long)(n0 + sn) * 64 + sd;
    const float* vp = vbuf + vbase + (long)(n0 + sn) * C_ + sd;
    v4f k0v = *(const v4f*)kp,       k1v = *(const v4f*)(kp + 4);
    v4f v0v = *(const v4f*)vp,       v1v = *(const v4f*)(vp + 4);
#pragma unroll
    for (int j = 0; j < 4; ++j) {
      sK[sd + j][sn]     = (_Float16)(expf(k0v[j] - smax[sd + j])     * sinv[sd + j]);
      sK[sd + 4 + j][sn] = (_Float16)(expf(k1v[j] - smax[sd + 4 + j]) * sinv[sd + 4 + j]);
      sV[sd + j][sn]     = (_Float16)v0v[j];
      sV[sd + 4 + j][sn] = (_Float16)v1v[j];
    }
    __syncthreads();
    v16h a  = frag_a(sK, rw * 16 + (lane & 15), lane);
    v16h f0 = frag_b(sV, cw * 32 + (lane & 15), lane);
    v16h f1 = frag_b(sV, cw * 32 + 16 + (lane & 15), lane);
    acc0 = WMMA_F16(a, f0, acc0);
    acc1 = WMMA_F16(a, f1, acc1);
    __syncthreads();
  }
  int nloc = lane & 15, mb = (lane < 16) ? 0 : 8;
#pragma unroll
  for (int r = 0; r < 8; ++r) {
    int drow = rw * 16 + mb + r;
    int e0 = cw * 32 + nloc;
    ctx[((long)hb * 64 + drow) * 64 + e0]      = acc0[r];
    ctx[((long)hb * 64 + drow) * 64 + e0 + 16] = acc1[r];
  }
}

// ---------------- row-wise L2 normalize (width 64) ----------------
__global__ __launch_bounds__(256) void l2norm_rows(float* __restrict__ X, long rows, float eps)
{
  __shared__ float red[256];
  long row = (long)blockIdx.x * 4 + (threadIdx.x >> 6);
  int  c   = threadIdx.x & 63;
  float x = 0.f;
  if (row < rows) x = X[row * 64 + c];
  red[threadIdx.x] = x * x;
  __syncthreads();
  for (int s = 32; s > 0; s >>= 1) {
    if (c < s) red[threadIdx.x] += red[threadIdx.x + s];
    __syncthreads();
  }
  float nrm = fmaxf(sqrtf(red[threadIdx.x & ~63]), eps);
  if (row < rows) X[row * 64 + c] = x / nrm;
}

// ---------------- contrastive loss partials (fixed-order, deterministic) ----
__global__ __launch_bounds__(256) void loss_partial(
    const float* __restrict__ simcat, const float* __restrict__ rcat,
    float* __restrict__ partials)
{
  int t = threadIdx.x;
  float acc = 0.f;
  for (long idx = (long)blockIdx.x * 256 + t; idx < (long)H_ * BN_;
       idx += (long)gridDim.x * 256) {
    const float* ss = simcat + idx * 32;   // [0..15]=spec, [16..31]=spat
    const float* rr = rcat   + idx * 32;
    float mxs = -1e30f, mxt = -1e30f, pos_s = 0.f, pos_t = 0.f;
#pragma unroll
    for (int m = 0; m < 16; ++m) {
      float a = ss[m], c = ss[16 + m];
      pos_s += a * rr[m]; pos_t += c * rr[16 + m];
      mxs = fmaxf(mxs, a); mxt = fmaxf(mxt, c);
    }
    float es = 0.f, et = 0.f;
#pragma unroll
    for (int m = 0; m < 16; ++m) {
      es += expf((ss[m] - mxs) * TAU_INV);
      et += expf((ss[16 + m] - mxt) * TAU_INV);
    }
    acc += pos_s * TAU_INV - (mxs * TAU_INV + logf(es));
    acc += pos_t * TAU_INV - (mxt * TAU_INV + logf(et));
  }
  __shared__ float red[256];
  red[t] = acc;
  __syncthreads();
  for (int s = 128; s > 0; s >>= 1) {
    if (t < s) red[t] += red[t + s];
    __syncthreads();
  }
  if (t == 0) partials[blockIdx.x] = red[0];
}

// ---------------- finalize: contrastive scalar + diversity scalar ----------
__global__ __launch_bounds__(256) void finalize_kernel(
    const float* __restrict__ partials,
    const float* __restrict__ Kspec, const float* __restrict__ Kspat,
    float* __restrict__ outs)
{
  int t = threadIdx.x;
  __shared__ float nrm[256];   // row norms: [0..127]=spec, [128..255]=spat
  __shared__ float red[256];
  {
    const float* P = (t < 128) ? Kspec : Kspat;
    int r = t & 127;
    float s = 0.f;
    for (int d = 0; d < 64; ++d) { float x = P[(long)r * 64 + d]; s += x * x; }
    nrm[t] = fmaxf(sqrtf(s), 1e-6f);
  }
  red[t] = partials[t];
  __syncthreads();
  for (int s = 128; s > 0; s >>= 1) {
    if (t < s) red[t] += red[t + s];
    __syncthreads();
  }
  if (t == 0) outs[0] = -0.5f * red[0] / (float)((long)H_ * BN_);
  __syncthreads();
  float dacc = 0.f;
  for (int p = t; p < 4096; p += 256) {
    int which = p >> 11, q = p & 2047;
    int h = q >> 8, m = (q >> 4) & 15, n = q & 15;
    if (m != n) {
      const float* P = which ? Kspat : Kspec;
      int rm = h * 16 + m, rn = h * 16 + n;
      float dot = 0.f;
      for (int d = 0; d < 64; ++d) dot += P[(long)rm * 64 + d] * P[(long)rn * 64 + d];
      dacc += dot / (nrm[which * 128 + rm] * nrm[which * 128 + rn]);
    }
  }
  red[t] = dacc;
  __syncthreads();
  for (int s = 128; s > 0; s >>= 1) {
    if (t < s) red[t] += red[t + s];
    __syncthreads();
  }
  if (t == 0) outs[1] = 0.01f * red[0] / (8.0f * 16.0f * 15.0f + 1e-6f);
}

// ---------------- host-side launcher ----------------
static inline void launch_gemm(hipStream_t s, const float* A, const float* Bm, float* Cm,
                               const float* bias, int M, int N, int K,
                               int lda, int ldb, int ldc,
                               long a1, long a2, long b1, long b2, long c1, long c2,
                               int nz, int zdiv, bool transB_guarded)
{
  dim3 g((N + 63) / 64, (M + 63) / 64, nz);
  if (transB_guarded)
    gemm_wmma<true, true><<<g, 256, 0, s>>>(A, Bm, Cm, bias, M, N, K, lda, ldb, ldc,
                                            a1, a2, b1, b2, c1, c2, zdiv);
  else
    gemm_wmma<false, false><<<g, 256, 0, s>>>(A, Bm, Cm, bias, M, N, K, lda, ldb, ldc,
                                              a1, a2, b1, b2, c1, c2, zdiv);
}

extern "C" void kernel_launch(void* const* d_in, const int* in_sizes, int n_in,
                              void* d_out, int out_size, void* d_ws, size_t ws_size,
                              hipStream_t stream)
{
  const float* x       = (const float*)d_in[0];
  const float* W_qkv   = (const float*)d_in[1];
  const float* W_proj  = (const float*)d_in[2];
  const float* b_proj  = (const float*)d_in[3];
  const float* W_rspec = (const float*)d_in[4];
  const float* b_rspec = (const float*)d_in[5];
  const float* W_rspat = (const float*)d_in[6];
  const float* b_rspat = (const float*)d_in[7];
  const float* K_spec  = (const float*)d_in[8];
  const float* K_spat  = (const float*)d_in[9];
  const float* W_tok   = (const float*)d_in[10];
  const float* b_tok   = (const float*)d_in[11];
  const float* W_prot  = (const float*)d_in[12];
  const float* b_prot  = (const float*)d_in[13];
  float* out = (float*)d_out;

  float* wsf = (float*)d_ws;
  size_t off = 0;
  auto alloc = [&](size_t n) { float* p = wsf + off; off += n; return p; };
  float* qbuf   = alloc((size_t)BN_ * C_);          // q projection (BN,512)
  float* vbuf   = alloc((size_t)BN_ * C_);          // v projection (BN,512)
  float* lgspec = alloc((size_t)BN_ * H_ * M_);     // router logits
  float* lgspat = alloc((size_t)BN_ * H_ * M_);
  float* rcat   = alloc((size_t)H_ * BN_ * 32);     // [r_spec|r_spat] (H,BN,32)
  float* kcat   = alloc((size_t)H_ * 32 * 64);      // [K_spec;K_spat] (H,32,64)
  float* keff   = alloc((size_t)H_ * BN_ * 64);     // K_eff (H,BN,64)
  float* mxb    = alloc((size_t)H_ * B_ * 64);
  float* smb    = alloc((size_t)H_ * B_ * 64);
  float* ctx    = alloc((size_t)H_ * B_ * 64 * 64); // context (H,B,64,64)
  float* attn   = alloc((size_t)BN_ * C_);          // q.ctx merged (BN,512)
  float* tbuf   = alloc((size_t)H_ * BN_ * 64);     // token embeddings
  float* pspec  = alloc((size_t)H_ * M_ * 64);
  float* pspat  = alloc((size_t)H_ * M_ * 64);
  float* pcat   = alloc((size_t)H_ * 32 * 64);      // [p_spec_n;p_spat_n]
  float* simcat = alloc((size_t)H_ * BN_ * 32);     // sims, matches rcat layout
  float* part   = alloc(256);
  (void)ws_size; (void)in_sizes; (void)n_in; (void)out_size;

  // 1) q and v projections (the k third of W_qkv is dead code in the reference)
  launch_gemm(stream, x, W_qkv,          qbuf, nullptr, BN_, C_, C_, C_, 3 * C_, C_,
              0, 0, 0, 0, 0, 0, 1, 1, false);
  launch_gemm(stream, x, W_qkv + 2 * C_, vbuf, nullptr, BN_, C_, C_, C_, 3 * C_, C_,
              0, 0, 0, 0, 0, 0, 1, 1, false);
  // 2) router logits (+bias)
  launch_gemm(stream, x, W_rspec, lgspec, b_rspec, BN_, H_ * M_, C_, C_, H_ * M_, H_ * M_,
              0, 0, 0, 0, 0, 0, 1, 1, false);
  launch_gemm(stream, x, W_rspat, lgspat, b_rspat, BN_, H_ * M_, C_, C_, H_ * M_, H_ * M_,
              0, 0, 0, 0, 0, 0, 1, 1, false);
  // 3) top-3 routing -> concatenated r (H,BN,32)
  routing_kernel<<<(H_ * BN_) / 256, 256, 0, stream>>>(lgspec, lgspat, rcat);
  // 4) bank concat + K_eff = rcat @ kcat (spec+spat folded into one K=32 GEMM)
  concat_kernel<<<(H_ * 32 * 64) / 256, 256, 0, stream>>>(K_spec, K_spat, kcat);
  launch_gemm(stream, rcat, kcat, keff, nullptr, BN_, 64, 32, 32, 64, 64,
              (long)BN_ * 32, 0, (long)32 * 64, 0, (long)BN_ * 64, 0, H_, 1, false);
  // 5) column softmax stats over N
  softmax_stats<<<H_ * B_, 256, 0, stream>>>(keff, mxb, smb);
  // 6) context = K_norm^T . v per (h,b)
  context_wmma<<<H_ * B_, 256, 0, stream>>>(keff, vbuf, mxb, smb, ctx);
  // 7) out_attn = q . ctx per (h,b), written into the (BN,512) merged buffer
  launch_gemm(stream, qbuf, ctx, attn, nullptr, N_, 64, 64, C_, 64, C_,
              64L, (long)N_ * C_,                 // A: +h*64 cols, +b*N rows
              (long)B_ * 64 * 64, (long)64 * 64,  // B: ctx[(h,b)]
              64L, (long)N_ * C_,                 // C: same pattern as A
              H_ * B_, B_, false);
  // 8) output projection (+bias) straight into d_out
  launch_gemm(stream, attn, W_proj, out, b_proj, BN_, C_, C_, C_, C_, C_,
              0, 0, 0, 0, 0, 0, 1, 1, false);
  // 9) token embeddings t = v @ W_tok + b_tok per head, then l2 normalize
  launch_gemm(stream, vbuf, W_tok, tbuf, b_tok, BN_, 64, 64, C_, 64, 64,
              64L, 0, 0, 0, (long)BN_ * 64, 0, H_, 1, false);
  l2norm_rows<<<(H_ * BN_) / 4, 256, 0, stream>>>(tbuf, (long)H_ * BN_, 1e-12f);
  // 10) prototypes p = K @ W_prot + b_prot, l2 normalized, then concatenated
  launch_gemm(stream, K_spec, W_prot, pspec, b_prot, H_ * M_, 64, 64, 64, 64, 64,
              0, 0, 0, 0, 0, 0, 1, 1, false);
  launch_gemm(stream, K_spat, W_prot, pspat, b_prot, H_ * M_, 64, 64, 64, 64, 64,
              0, 0, 0, 0, 0, 0, 1, 1, false);
  l2norm_rows<<<(H_ * M_ + 3) / 4, 256, 0, stream>>>(pspec, (long)H_ * M_, 1e-12f);
  l2norm_rows<<<(H_ * M_ + 3) / 4, 256, 0, stream>>>(pspat, (long)H_ * M_, 1e-12f);
  concat_kernel<<<(H_ * 32 * 64) / 256, 256, 0, stream>>>(pspec, pspat, pcat);
  // 11) simcat = t . [p_spec|p_spat]^T per head (N=32, B transposed, guarded)
  launch_gemm(stream, tbuf, pcat, simcat, nullptr, BN_, 32, 64, 64, 64, 32,
              (long)BN_ * 64, 0, (long)32 * 64, 0, (long)BN_ * 32, 0, H_, 1, true);
  // 12) deterministic two-stage loss reduction + diversity
  loss_partial<<<256, 256, 0, stream>>>(simcat, rcat, part);
  finalize_kernel<<<1, 256, 0, stream>>>(part, K_spec, K_spat, out + (size_t)BN_ * C_);
}
